// AxialAttention_38757784879712
// MI455X (gfx1250) — matrix operand order repaired and checked by reading, not compiled, not run on hardware
//
#include <hip/hip_runtime.h>
#include <cstdint>

// Problem constants (match reference)
#define S   128
#define DM  256
#define NH  8
#define HD  32
#define SCALE_F 0.17677669529663687f   // 32^-0.5
#define MASKB  (-1.0e9f)

typedef __bf16 bf16;
typedef __attribute__((ext_vector_type(16))) __bf16 v16bf;
typedef __attribute__((ext_vector_type(8)))  __bf16 v8bf;
typedef __attribute__((ext_vector_type(8)))  float  v8f;

// ---- LDS layout (bytes). Pitches padded to dodge bank conflicts. ----
#define QP   264   // q / out buffer pitch (bf16 elems per row), rows = 128
#define VTP  136   // v-transposed pitch, rows = 256 (channel-major)
#define AP   136   // attn pitch, rows = 128
#define WP   264   // W staging pitch, rows = 16 per slot
#define WSLOT_E (16 * WP)                      // elements per W slot
#define OFF_Q    0
#define OFF_K    (OFF_Q    + 128 * QP  * 2)    //  67584
#define OFF_VT   (OFF_K    + 128 * QP  * 2)    // 135168 (xn aliases this region)
#define OFF_ATTN (OFF_VT   + 256 * VTP * 2)    // 204800
#define OFF_W    (OFF_ATTN + 128 * AP  * 2)    // 239616
#define SMEM_BYTES (OFF_W  + 2 * WSLOT_E * 2)  // 256512 (~250 KB of 320 KB WGP LDS)

static __device__ __forceinline__ bf16 tobf(float f) {
  return (bf16)f;   // native f32->bf16 convert (RNE) on gfx1250
}

static __device__ __forceinline__ v16bf ld_frag(const bf16* lo8, const bf16* hi8) {
  v8bf lo = *(const v8bf*)lo8;
  v8bf hi = *(const v8bf*)hi8;
  return __builtin_shufflevector(lo, hi, 0,1,2,3,4,5,6,7,8,9,10,11,12,13,14,15);
}

static __device__ __forceinline__ v8f wmma_bf16(v16bf a, v16bf b, v8f c) {
  // 8-arg form: (neg_a, A, neg_b, B, c_mod, C, reuse_a, reuse_b)
  return __builtin_amdgcn_wmma_f32_16x16x32_bf16(false, a, false, b, (short)0, c,
                                                 false, false);
}

#define V8F_ZERO {0.f,0.f,0.f,0.f,0.f,0.f,0.f,0.f}

// Cooperative stage of 16 weight rows [16t,16t+16) x 256 cols as bf16 into one W slot.
static __device__ __forceinline__ void stage_w(bf16* slot, const float* __restrict__ W,
                                               int t, int tid) {
  const int n  = tid >> 4;
  const int d0 = (tid & 15) << 4;
  const float4* src = (const float4*)(W + (size_t)(t * 16 + n) * DM + d0);
  bf16* dst = slot + n * WP + d0;
  #pragma unroll
  for (int q4 = 0; q4 < 4; ++q4) {
    float4 f = src[q4];
    dst[q4 * 4 + 0] = tobf(f.x); dst[q4 * 4 + 1] = tobf(f.y);
    dst[q4 * 4 + 2] = tobf(f.z); dst[q4 * 4 + 3] = tobf(f.w);
  }
}

__global__ void __launch_bounds__(256)
axial_attn_wmma_kernel(const float* __restrict__ pair_act,
                       const int*   __restrict__ pair_mask,
                       const float* __restrict__ ln_gamma,
                       const float* __restrict__ ln_beta,
                       const float* __restrict__ Wqkv,
                       const float* __restrict__ Wout,
                       float* __restrict__ out)
{
  extern __shared__ char smem[];
  bf16* qbuf    = (bf16*)(smem + OFF_Q);     // 128 x QP : q channels -> later attn-out
  bf16* kbuf    = (bf16*)(smem + OFF_K);     // 128 x QP : k channels
  bf16* vtbuf   = (bf16*)(smem + OFF_VT);    // 256 x VTP: v transposed (chan-major)
  bf16* xnbuf   = (bf16*)(smem + OFF_VT);    // alias: 128 x QP normalized input
  bf16* attnbuf = (bf16*)(smem + OFF_ATTN);  // 128 x AP : attention probabilities
  bf16* wbuf    = (bf16*)(smem + OFF_W);     // 2 x (16 x WP) : double-buffered W tile

  const int tid   = threadIdx.x;
  const int wave  = tid >> 5;
  const int lane  = tid & 31;
  const int laneN = lane & 15;     // column / row-in-tile index
  const int half  = lane >> 4;     // upper/lower half of wave
  const int m0    = wave * 16;     // this wave's row strip [m0, m0+16)
  const int x     = blockIdx.x;
  const int b     = blockIdx.y;

  const size_t rowbase = (size_t)(b * S + x) * S;
  const float* xin = pair_act + rowbase * DM;
  const int*   pm  = pair_mask + rowbase;

  // ---------------- Phase 0: LayerNorm -> xnbuf (bf16), wave-private rows ----------
  {
    float gam[8], bet[8];
    #pragma unroll
    for (int j = 0; j < 8; ++j) {
      gam[j] = ln_gamma[lane * 8 + j];
      bet[j] = ln_beta [lane * 8 + j];
    }
    #pragma unroll 1
    for (int i = 0; i < 16; ++i) {
      const int y = m0 + i;
      const float4* p = (const float4*)(xin + (size_t)y * DM + lane * 8);
      float4 f0 = p[0], f1 = p[1];
      float v[8] = {f0.x, f0.y, f0.z, f0.w, f1.x, f1.y, f1.z, f1.w};
      float s1 = 0.f, s2 = 0.f;
      #pragma unroll
      for (int j = 0; j < 8; ++j) { s1 += v[j]; s2 += v[j] * v[j]; }
      #pragma unroll
      for (int off = 16; off; off >>= 1) {
        s1 += __shfl_xor(s1, off);
        s2 += __shfl_xor(s2, off);
      }
      const float mu  = s1 * (1.f / 256.f);
      const float var = s2 * (1.f / 256.f) - mu * mu;
      const float rs  = rsqrtf(var + 1e-5f);
      bf16* dst = xnbuf + (size_t)y * QP + lane * 8;
      #pragma unroll
      for (int j = 0; j < 8; ++j)
        dst[j] = tobf((v[j] - mu) * rs * gam[j] + bet[j]);
    }
  }
  // xn strip is wave-private; A-fragments read our own rows -> no barrier needed.

  // ---------------- Phase 1: QKV GEMM (128x256 @ 256x768), rotary, store q/k/vT ----
  v16bf afrag[8];
  {
    const bf16* arow = xnbuf + (size_t)(m0 + laneN) * QP;
    #pragma unroll
    for (int kk = 0; kk < 8; ++kk)
      afrag[kk] = ld_frag(arow + 32 * kk + 8 * half,
                          arow + 32 * kk + 16 + 8 * half);
  }
  const float tlx = -1.f + (2.f / 127.f) * (float)x;   // linspace(-1,1,128)[x]
  // inv_freq[j] = 10000^(-j/8); j = laneN>>1
  const float invf = exp2f(-(float)(laneN >> 1) * (0.125f * 13.287712379549449f));
  const float rotsgn = (laneN & 1) ? 1.f : -1.f;

  stage_w(wbuf, Wqkv, 0, tid);   // prologue stage of tile 0 into slot 0
  #pragma unroll 1
  for (int t = 0; t < 48; ++t) {
    __syncthreads();   // tile t staged & visible; previous tile's readers done
    if (t + 1 < 48) stage_w(wbuf + ((t + 1) & 1) * WSLOT_E, Wqkv, t + 1, tid);
    const bf16* wb = wbuf + (t & 1) * WSLOT_E + laneN * WP + 8 * half;

    // preload all 8 B fragments, then 8 back-to-back WMMAs (accumulator chain)
    v16bf bfr[8];
    #pragma unroll
    for (int kk = 0; kk < 8; ++kk)
      bfr[kk] = ld_frag(wb + 32 * kk, wb + 32 * kk + 16);
    v8f acc = V8F_ZERO;
    #pragma unroll
    for (int kk = 0; kk < 8; ++kk)
      acc = wmma_bf16(afrag[kk], bfr[kk], acc);

    // Rotary on q channels 0..31 (t==0,1) and k channels 0..31 (t==16,17).
    // Pairs are adjacent columns == adjacent lanes -> shfl_xor(1).
    if (t == 0 || t == 1 || t == 16 || t == 17) {
      const bool waxis = (t & 1);   // tile of channels 16..31 uses f[y], else f[x]
      #pragma unroll
      for (int r = 0; r < 8; ++r) {
        const float prt  = __shfl_xor(acc[r], 1);
        const float base = waxis ? (-1.f + (2.f / 127.f) * (float)(m0 + 8 * half + r))
                                 : tlx;
        float sn, cs;
        __sincosf(base * invf, &sn, &cs);
        acc[r] = acc[r] * cs + rotsgn * prt * sn;
      }
    }

    // C layout: lane holds column laneN, element r -> row m0 + 8*half + r
    if (t < 16) {            // q channels [16t, 16t+16)
      #pragma unroll
      for (int r = 0; r < 8; ++r)
        qbuf[(size_t)(m0 + 8 * half + r) * QP + 16 * t + laneN] = tobf(acc[r]);
    } else if (t < 32) {     // k channels
      #pragma unroll
      for (int r = 0; r < 8; ++r)
        kbuf[(size_t)(m0 + 8 * half + r) * QP + 16 * (t - 16) + laneN] = tobf(acc[r]);
    } else {                 // v, stored transposed: vtbuf[chan][y]
      #pragma unroll
      for (int r = 0; r < 8; ++r)
        vtbuf[(size_t)(16 * (t - 32) + laneN) * VTP + (m0 + 8 * half + r)] = tobf(acc[r]);
    }
  }
  __syncthreads();   // q/k/vT now visible to all waves

  // ---------------- Phase 2: per-head attention (barrier-free: strips wave-private)
  #pragma unroll 1
  for (int h = 0; h < NH; ++h) {
    // A = q strip (16x32), one fragment; K = HEAD_DIM = 32 -> single WMMA per tile
    v16bf aq;
    {
      const bf16* qrow = qbuf + (size_t)(m0 + laneN) * QP + 32 * h;
      aq = ld_frag(qrow + 8 * half, qrow + 16 + 8 * half);
    }
    // preload all 8 K^T fragments, then 8 independent WMMAs (no D->A/B hazard)
    v16bf bk[8];
    #pragma unroll
    for (int t = 0; t < 8; ++t) {
      const bf16* krow = kbuf + (size_t)(16 * t + laneN) * QP + 32 * h + 8 * half;
      bk[t] = ld_frag(krow, krow + 16);
    }
    float sc[8][8];
    #pragma unroll
    for (int t = 0; t < 8; ++t) {
      v8f c = V8F_ZERO;
      c = wmma_bf16(aq, bk[t], c);
      const int masked = pm[16 * t + laneN];   // mask depends on (b,x,kpos) only
      #pragma unroll
      for (int r = 0; r < 8; ++r)
        sc[t][r] = masked ? MASKB : c[r] * SCALE_F;
    }
    // softmax over 128 keys: reduce across 8 tiles + 16 lanes of this half
    #pragma unroll
    for (int r = 0; r < 8; ++r) {
      float mx = sc[0][r];
      #pragma unroll
      for (int t = 1; t < 8; ++t) mx = fmaxf(mx, sc[t][r]);
      #pragma unroll
      for (int off = 8; off; off >>= 1) mx = fmaxf(mx, __shfl_xor(mx, off));
      float sum = 0.f;
      #pragma unroll
      for (int t = 0; t < 8; ++t) { float e = __expf(sc[t][r] - mx); sc[t][r] = e; sum += e; }
      #pragma unroll
      for (int off = 8; off; off >>= 1) sum += __shfl_xor(sum, off);
      const float inv = 1.f / sum;
      #pragma unroll
      for (int t = 0; t < 8; ++t) sc[t][r] *= inv;
    }
    // spill attn strip (wave-private rows) for the cross-lane A-layout transpose
    #pragma unroll
    for (int t = 0; t < 8; ++t)
      #pragma unroll
      for (int r = 0; r < 8; ++r)
        attnbuf[(size_t)(m0 + 8 * half + r) * AP + 16 * t + laneN] = tobf(sc[t][r]);

    // out_strip(16x32) = attn(16x128) @ v_head(128x32); write into dead q cols of h
    #pragma unroll
    for (int dt = 0; dt < 2; ++dt) {
      v16bf aa[4], bb[4];
      #pragma unroll
      for (int kk = 0; kk < 4; ++kk) {
        const bf16* arow = attnbuf + (size_t)(m0 + laneN) * AP + 32 * kk;
        aa[kk] = ld_frag(arow + 8 * half, arow + 16 + 8 * half);
        const bf16* brow = vtbuf + (size_t)(32 * h + 16 * dt + laneN) * VTP
                         + 32 * kk + 8 * half;
        bb[kk] = ld_frag(brow, brow + 16);
      }
      v8f oc = V8F_ZERO;
      #pragma unroll
      for (int kk = 0; kk < 4; ++kk)
        oc = wmma_bf16(aa[kk], bb[kk], oc);
      #pragma unroll
      for (int r = 0; r < 8; ++r)
        qbuf[(size_t)(m0 + 8 * half + r) * QP + 32 * h + 16 * dt + laneN] = tobf(oc[r]);
    }
  }

  // ---------------- Phase 3: output projection (128x256 @ 256x256) -> global -------
  v16bf ofrag[8];
  {
    const bf16* orow = qbuf + (size_t)(m0 + laneN) * QP;   // own strip, own writes
    #pragma unroll
    for (int kk = 0; kk < 8; ++kk)
      ofrag[kk] = ld_frag(orow + 32 * kk + 8 * half,
                          orow + 32 * kk + 16 + 8 * half);
  }
  float* outp = out + rowbase * DM;
  stage_w(wbuf, Wout, 0, tid);   // slot 0 free: all waves past phase-1 barrier
  #pragma unroll 1
  for (int t = 0; t < 16; ++t) {
    __syncthreads();   // tile t staged; previous readers done
    if (t + 1 < 16) stage_w(wbuf + ((t + 1) & 1) * WSLOT_E, Wout, t + 1, tid);
    const bf16* wb = wbuf + (t & 1) * WSLOT_E + laneN * WP + 8 * half;

    v16bf bfr[8];
    #pragma unroll
    for (int kk = 0; kk < 8; ++kk)
      bfr[kk] = ld_frag(wb + 32 * kk, wb + 32 * kk + 16);
    v8f acc = V8F_ZERO;
    #pragma unroll
    for (int kk = 0; kk < 8; ++kk)
      acc = wmma_bf16(ofrag[kk], bfr[kk], acc);

    #pragma unroll
    for (int r = 0; r < 8; ++r)
      outp[(size_t)(m0 + 8 * half + r) * DM + 16 * t + laneN] = acc[r];
  }
}

extern "C" void kernel_launch(void* const* d_in, const int* in_sizes, int n_in,
                              void* d_out, int out_size, void* d_ws, size_t ws_size,
                              hipStream_t stream) {
  (void)in_sizes; (void)n_in; (void)d_ws; (void)ws_size; (void)out_size;
  const float* pair_act = (const float*)d_in[0];
  const int*   pair_mask= (const int*)  d_in[1];
  const float* ln_gamma = (const float*)d_in[2];
  const float* ln_beta  = (const float*)d_in[3];
  const float* Wqkv     = (const float*)d_in[4];
  const float* Wout     = (const float*)d_in[5];
  float* outp = (float*)d_out;

  // ~250 KB dynamic LDS (WGP has 320 KB); idempotent, deterministic.
  (void)hipFuncSetAttribute((const void*)axial_attn_wmma_kernel,
                            hipFuncAttributeMaxDynamicSharedMemorySize, SMEM_BYTES);

  // one workgroup per (b, x) row: grid = (128, 2), 256 threads = 8 waves (wave32)
  axial_attn_wmma_kernel<<<dim3(S, 2, 1), dim3(256, 1, 1), SMEM_BYTES, stream>>>(
      pair_act, pair_mask, ln_gamma, ln_beta, Wqkv, Wout, outp);
}
